// EGATModel_41893111005430
// MI455X (gfx1250) — matrix operand order repaired and checked by reading, not compiled, not run on hardware
//
#include <hip/hip_runtime.h>
#include <hip/hip_bf16.h>

#define N_NODES 10000
#define N_EDGES 160000
#define HDIM    128
#define NHEADS  4
#define HHDIM   512
#define NLAYERS 2

typedef __attribute__((ext_vector_type(16))) _Float16 v16h;
typedef __attribute__((ext_vector_type(4)))  _Float16 v4h;
typedef __attribute__((ext_vector_type(8)))  float    v8f;

__device__ __forceinline__ unsigned f2ord(float f) {
    unsigned u = __float_as_uint(f);
    return (u & 0x80000000u) ? ~u : (u | 0x80000000u);
}
__device__ __forceinline__ float ord2f(unsigned u) {
    return __uint_as_float((u & 0x80000000u) ? (u & 0x7FFFFFFFu) : ~u);
}

// Pack f32 weight matrix [K x Ncols] into WMMA B-fragment tile order:
// for K-tile kt, N-tile nt, lane holds col=lane&15, K-range (lane>>4)*16..+15,
// stored as 16 contiguous halves at ((kt*NT + nt)*32 + lane)*16.
__global__ void pack_w_f16(const float* __restrict__ in, _Float16* __restrict__ out,
                           int K, int Ncols) {
    int idx = blockIdx.x * blockDim.x + threadIdx.x; // tile*32 + lane
    int total = (K >> 5) * (Ncols >> 4) * 32;
    if (idx >= total) return;
    int lane = idx & 31;
    int tile = idx >> 5;
    int NT = Ncols >> 4;
    int kt = tile / NT, nt = tile - kt * NT;
    int col = lane & 15, hi = lane >> 4;
    const float* src = in + (size_t)(kt * 32 + hi * 16) * Ncols + nt * 16 + col;
    _Float16* dst = out + (size_t)idx * 16;
#pragma unroll
    for (int i = 0; i < 16; ++i) dst[i] = (_Float16)src[(size_t)i * Ncols];
}

__global__ void zero_f32(float* __restrict__ p, int n) {
    int i = blockIdx.x * blockDim.x + threadIdx.x;
    if (i < n) p[i] = 0.0f;
}

// C[M x NC] = A[M x K] @ B[K x NC], B pre-packed in fragment order.
// EPI 0: plain store
// EPI 1: v += f_ni[src[row]][col] + f_nj[dst[row]][col] + bias[col]; leaky_relu(0.01);
//        also accumulates scores[row][head] += v * attn[col] (fused attention dot)
// EPI 2: v += bias[col]; elu; v += residual[row][col]
template <int EPI, int K, int NC>
__global__ __launch_bounds__(256) void gemm_wmma(
    const float* __restrict__ A,
    const _Float16* __restrict__ Bp,
    float* __restrict__ C,
    const float* __restrict__ gSrcFeat,
    const float* __restrict__ gDstFeat,
    const int* __restrict__ srcIdx,
    const int* __restrict__ dstIdx,
    const float* __restrict__ bias,
    const float* __restrict__ residual,
    const float* __restrict__ attnW,
    float* __restrict__ scoresOut)
{
    constexpr int NT = NC / 16;
    __shared__ _Float16 ldsA[16 * K];
    __shared__ int s_src[16], s_dst[16];

    const int m0  = blockIdx.x * 16;
    const int tid = threadIdx.x;

    // stage 16 x K strip of A (contiguous since lda==K) into LDS as f16, float4 at a time
    {
        const float* Ab = A + (size_t)m0 * K;
        for (int i = tid * 4; i < 16 * K; i += 256 * 4) {
            float4 f = *(const float4*)(Ab + i);
            v4h h; h[0] = (_Float16)f.x; h[1] = (_Float16)f.y;
                   h[2] = (_Float16)f.z; h[3] = (_Float16)f.w;
            *(v4h*)&ldsA[i] = h;
        }
    }
    if (EPI == 1 && tid < 16) {
        s_src[tid] = srcIdx[m0 + tid];
        s_dst[tid] = dstIdx[m0 + tid];
    }
    __syncthreads();

    const int wave = tid >> 5;
    const int lane = tid & 31;
    const int n0   = wave * 64;
    if (n0 >= NC) return;
    const int col = lane & 15;
    const int hi  = lane >> 4;

    v8f acc[4] = {{}, {}, {}, {}};

    for (int k0 = 0; k0 < K; k0 += 32) {
        // A fragment 16x32, contiguous K order: lane half hi holds k0+hi*16..+15
        v16h a;
        {
            const _Float16* ap = &ldsA[(lane & 15) * K + k0 + hi * 16];
#pragma unroll
            for (int i = 0; i < 8; ++i) { a[i] = ap[i]; a[8 + i] = ap[8 + i]; }
        }
        const _Float16* bbase =
            Bp + ((size_t)((k0 >> 5) * NT + (n0 >> 4)) * 32 + lane) * 16;
#pragma unroll
        for (int t = 0; t < 4; ++t) {
            v16h b = *(const v16h*)(bbase + (size_t)t * (32 * 16));
            acc[t] = __builtin_amdgcn_wmma_f32_16x16x32_f16(
                false, a, false, b, (short)0, acc[t], false, false);
        }
    }

    float av[4];
    float p[8];
    if (EPI == 1) {
#pragma unroll
        for (int t = 0; t < 4; ++t) av[t] = attnW[n0 + t * 16 + col];
#pragma unroll
        for (int r = 0; r < 8; ++r) p[r] = 0.0f;
    }

    // C/D layout: lane(0-15) N=lane, VGPR r -> M=r; lane(16-31) N=lane-16, M=8+r
#pragma unroll
    for (int t = 0; t < 4; ++t) {
#pragma unroll
        for (int r = 0; r < 8; ++r) {
            const int lrow = hi * 8 + r;
            const int row  = m0 + lrow;
            const int cg   = n0 + t * 16 + col;
            float v = acc[t][r];
            if (EPI == 0) {
                C[(size_t)row * NC + cg] = v;
            } else if (EPI == 1) {
                v += gSrcFeat[(size_t)s_src[lrow] * HHDIM + cg]
                   + gDstFeat[(size_t)s_dst[lrow] * HHDIM + cg]
                   + bias[cg];
                v = v > 0.0f ? v : 0.01f * v;
                C[(size_t)row * NC + cg] = v;
                p[r] += v * av[t];
            } else {
                v += bias[cg];
                v = v > 0.0f ? v : (__expf(v) - 1.0f);
                v += residual[(size_t)row * NC + cg];
                C[(size_t)row * NC + cg] = v;
            }
        }
    }

    if (EPI == 1) {
        // reduce partial scores across the 16 lanes of each half-wave
#pragma unroll
        for (int r = 0; r < 8; ++r) {
            float s = p[r];
            s += __shfl_xor(s, 8, 32);
            s += __shfl_xor(s, 4, 32);
            s += __shfl_xor(s, 2, 32);
            s += __shfl_xor(s, 1, 32);
            p[r] = s;
        }
        if ((lane & 15) == 0) {
            const int head = n0 >> 7; // 64-col wave maps to a single head
#pragma unroll
            for (int r = 0; r < 8; ++r)
                unsafeAtomicAdd(&scoresOut[(size_t)(m0 + hi * 8 + r) * NHEADS + head], p[r]);
        }
    }
}

__global__ void segmax_kernel(const float* __restrict__ scores,
                              const int* __restrict__ dst,
                              unsigned* __restrict__ m_u, int nE)
{
    int i = blockIdx.x * blockDim.x + threadIdx.x;
    if (i >= nE * NHEADS) return;
    int e = i >> 2, head = i & 3;
    atomicMax(&m_u[dst[e] * NHEADS + head], f2ord(scores[i]));
}

__global__ void expsum_kernel(const float* __restrict__ scores,
                              const unsigned* __restrict__ m_u,
                              const int* __restrict__ dst,
                              float* __restrict__ ex, float* __restrict__ denom, int nE)
{
    int i = blockIdx.x * blockDim.x + threadIdx.x;
    if (i >= nE * NHEADS) return;
    int e = i >> 2, head = i & 3;
    int d = dst[e];
    float m = ord2f(m_u[d * NHEADS + head]);
    float v = __expf(scores[i] - m);
    ex[i] = v;
    unsafeAtomicAdd(&denom[d * NHEADS + head], v);
}

// block per edge: h_out[dst] += h[src] * a
__global__ __launch_bounds__(256) void agg_kernel(
    const float* __restrict__ h, const float* __restrict__ ex,
    const float* __restrict__ denom, const int* __restrict__ src,
    const int* __restrict__ dst, float* __restrict__ h_out)
{
    int e = blockIdx.x;
    int s = src[e], d = dst[e];
    for (int c = threadIdx.x; c < HHDIM; c += blockDim.x) {
        int head = c >> 7;
        float a = ex[e * NHEADS + head] / denom[d * NHEADS + head];
        unsafeAtomicAdd(&h_out[(size_t)d * HHDIM + c], h[(size_t)s * HHDIM + c] * a);
    }
}

extern "C" void kernel_launch(void* const* d_in, const int* in_sizes, int n_in,
                              void* d_out, int out_size, void* d_ws, size_t ws_size,
                              hipStream_t stream) {
    (void)in_sizes; (void)n_in; (void)out_size; (void)ws_size;

    const float* nfeats  = (const float*)d_in[0];
    const float* efeats  = (const float*)d_in[1];
    const int*   src     = (const int*)d_in[2];
    const int*   dst     = (const int*)d_in[3];
    const float* W_ni    = (const float*)d_in[4];
    const float* W_nj    = (const float*)d_in[5];
    const float* W_fij   = (const float*)d_in[6];
    const float* attn    = (const float*)d_in[7];
    const float* bias_e  = (const float*)d_in[8];
    const float* W_node  = (const float*)d_in[9];
    const float* W_mlp_n = (const float*)d_in[10];
    const float* b_mlp_n = (const float*)d_in[11];
    const float* W_mlp_e = (const float*)d_in[12];
    const float* b_mlp_e = (const float*)d_in[13];

    char* ws = (char*)d_ws;
    size_t off = 0;
    auto alloc = [&](size_t bytes) -> void* {
        void* p = ws + off;
        off = (off + bytes + 255) & ~(size_t)255;
        return p;
    };

    const size_t WELEMS = (size_t)HDIM * HHDIM; // 65536
    _Float16* wh[NLAYERS][6];
    for (int l = 0; l < NLAYERS; ++l)
        for (int m = 0; m < 6; ++m)
            wh[l][m] = (_Float16*)alloc(WELEMS * sizeof(_Float16));

    float*    f_ni   = (float*)alloc((size_t)N_NODES * HHDIM * 4);
    float*    f_nj   = (float*)alloc((size_t)N_NODES * HHDIM * 4);
    float*    hbuf   = (float*)alloc((size_t)N_NODES * HHDIM * 4);
    // contiguous zero region: m_u | denom | scores | h_out (each multiple of 256B)
    unsigned* m_u    = (unsigned*)alloc((size_t)N_NODES * NHEADS * 4);
    float*    denom  = (float*)alloc((size_t)N_NODES * NHEADS * 4);
    float*    scores = (float*)alloc((size_t)N_EDGES * NHEADS * 4);
    float*    h_out  = (float*)alloc((size_t)N_NODES * HHDIM * 4);
    float*    f_out  = (float*)alloc((size_t)N_EDGES * HHDIM * 4);
    float*    exv    = (float*)alloc((size_t)N_EDGES * NHEADS * 4);
    float*    n_tmp  = (float*)alloc((size_t)N_NODES * HDIM * 4);
    float*    e_tmp  = (float*)alloc((size_t)N_EDGES * HDIM * 4);

    // pack weights into f16 fragment order
    // order: W_ni, W_nj, W_fij, W_node (128x512), W_mlp_n, W_mlp_e (512x128)
    const float* wsrcs[6] = {W_ni, W_nj, W_fij, W_node, W_mlp_n, W_mlp_e};
    const int    wK[6]    = {HDIM, HDIM, HDIM, HDIM, HHDIM, HHDIM};
    const int    wN[6]    = {HHDIM, HHDIM, HHDIM, HHDIM, HDIM, HDIM};
    for (int l = 0; l < NLAYERS; ++l)
        for (int m = 0; m < 6; ++m) {
            int total = (wK[m] >> 5) * (wN[m] >> 4) * 32;
            pack_w_f16<<<(total + 255) / 256, 256, 0, stream>>>(
                wsrcs[m] + l * WELEMS, wh[l][m], wK[m], wN[m]);
        }

    for (int l = 0; l < NLAYERS; ++l) {
        const float* nf_in  = (l == 0) ? nfeats : n_tmp;
        const float* ef_in  = (l == 0) ? efeats : e_tmp;
        float* nf_out = (l == NLAYERS - 1) ? (float*)d_out : n_tmp;
        float* ef_out = (l == NLAYERS - 1) ? (float*)d_out + (size_t)N_NODES * HDIM : e_tmp;

        // node GEMMs: f_ni, f_nj, h
        gemm_wmma<0, HDIM, HHDIM><<<N_NODES / 16, 256, 0, stream>>>(
            nf_in, wh[l][0], f_ni, nullptr, nullptr, nullptr, nullptr,
            nullptr, nullptr, nullptr, nullptr);
        gemm_wmma<0, HDIM, HHDIM><<<N_NODES / 16, 256, 0, stream>>>(
            nf_in, wh[l][1], f_nj, nullptr, nullptr, nullptr, nullptr,
            nullptr, nullptr, nullptr, nullptr);
        gemm_wmma<0, HDIM, HHDIM><<<N_NODES / 16, 256, 0, stream>>>(
            nf_in, wh[l][3], hbuf, nullptr, nullptr, nullptr, nullptr,
            nullptr, nullptr, nullptr, nullptr);

        // zero m_u, denom, scores, h_out (contiguous, before fused edge GEMM)
        {
            int zn = N_NODES * NHEADS * 2 + N_EDGES * NHEADS + N_NODES * HHDIM;
            zero_f32<<<(zn + 255) / 256, 256, 0, stream>>>((float*)m_u, zn);
        }

        // edge GEMM + gather + leaky_relu -> f_out, fused attention scores
        gemm_wmma<1, HDIM, HHDIM><<<N_EDGES / 16, 256, 0, stream>>>(
            ef_in, wh[l][2], f_out, f_ni, f_nj, src, dst,
            bias_e + (size_t)l * HHDIM, nullptr,
            attn + (size_t)l * NHEADS * HDIM, scores);

        segmax_kernel<<<(N_EDGES * NHEADS + 255) / 256, 256, 0, stream>>>(scores, dst, m_u, N_EDGES);
        expsum_kernel<<<(N_EDGES * NHEADS + 255) / 256, 256, 0, stream>>>(scores, m_u, dst, exv, denom, N_EDGES);
        agg_kernel<<<N_EDGES, 256, 0, stream>>>(hbuf, exv, denom, src, dst, h_out);

        // node MLP: elu(h_out @ W_mlp_n + b) + n
        gemm_wmma<2, HHDIM, HDIM><<<N_NODES / 16, 256, 0, stream>>>(
            h_out, wh[l][4], nf_out, nullptr, nullptr, nullptr, nullptr,
            b_mlp_n + (size_t)l * HDIM, nf_in, nullptr, nullptr);

        // edge MLP: elu(f_out @ W_mlp_e + b) + e
        gemm_wmma<2, HHDIM, HDIM><<<N_EDGES / 16, 256, 0, stream>>>(
            f_out, wh[l][5], ef_out, nullptr, nullptr, nullptr, nullptr,
            b_mlp_e + (size_t)l * HDIM, ef_in, nullptr, nullptr);
    }
}
